// CustomLlamaAttention_84765474554560
// MI455X (gfx1250) — compile-verified, matrix operand order checked by reading, not run on hardware
//
#include <hip/hip_runtime.h>
#include <hip/hip_bf16.h>

// Problem constants (from reference)
#define B_   2
#define S_   2048
#define H_   2048
#define NH_  32
#define NKV_ 8
#define HD_  64
#define BS_  (B_ * S_)      // 4096 tokens
#define KV_  (NKV_ * HD_)   // 512

typedef __attribute__((ext_vector_type(16))) __bf16 v16bf;
typedef __attribute__((ext_vector_type(8)))  float  v8f;

union FragU { v16bf v; uint4 u[2]; };

// Load one 16-bit WMMA fragment half-row: per-lane K packing is
// elems 0..7 = K khalf..khalf+7, elems 8..15 = K khalf+16..khalf+23.
__device__ __forceinline__ v16bf load_frag(const __hip_bfloat16* p, int khalf) {
    FragU f;
    f.u[0] = *reinterpret_cast<const uint4*>(p + khalf);
    f.u[1] = *reinterpret_cast<const uint4*>(p + khalf + 16);
    return f.v;
}

__device__ __forceinline__ v8f wmma_bf16(v16bf a, v16bf b, v8f c) {
    return __builtin_amdgcn_wmma_f32_16x16x32_bf16(false, a, false, b, (short)0, c, false, false);
}

// 16-lane (DPP row) max butterfly: xor1, xor2 via quad_perm, then row_ror 4/8.
// Stays within each 16-lane row => correct for the wave32 C/D layout halves.
__device__ __forceinline__ float rowmax16_dpp(float x) {
    union { float f; int i; } u, v;
    u.f = x;
    v.i = __builtin_amdgcn_update_dpp(u.i, u.i, 0xB1,  0xF, 0xF, false); x = fmaxf(x, v.f); u.f = x;
    v.i = __builtin_amdgcn_update_dpp(u.i, u.i, 0x4E,  0xF, 0xF, false); x = fmaxf(x, v.f); u.f = x;
    v.i = __builtin_amdgcn_update_dpp(u.i, u.i, 0x124, 0xF, 0xF, false); x = fmaxf(x, v.f); u.f = x;
    v.i = __builtin_amdgcn_update_dpp(u.i, u.i, 0x128, 0xF, 0xF, false); x = fmaxf(x, v.f);
    return x;
}

template <typename T> __device__ __forceinline__ T to_out(float x);
template <> __device__ __forceinline__ float to_out<float>(float x) { return x; }
template <> __device__ __forceinline__ __hip_bfloat16 to_out<__hip_bfloat16>(float x) { return __float2bfloat16(x); }

// ---------------------------------------------------------------------------
// Codebook gather -> transposed bf16 weight  WT[n*K + k] = W[k][n]
// ---------------------------------------------------------------------------
__global__ void gather_wT_kernel(const float* __restrict__ bank, const int* __restrict__ idx,
                                 __hip_bfloat16* __restrict__ WT, int Kin, int Nout) {
    const size_t t = (size_t)blockIdx.x * blockDim.x + threadIdx.x;
    const size_t total = (size_t)Kin * Nout;
    if (t >= total) return;
    const int n = (int)(t / Kin);
    const int k = (int)(t % Kin);
    const size_t f = (size_t)k * Nout + n;          // flat index in [Kin, Nout] row-major
    const int row = idx[f >> 3];
    WT[t] = __float2bfloat16(bank[(size_t)row * 8 + (f & 7)]);
}

__global__ void cast_bf16_kernel(const float* __restrict__ in, __hip_bfloat16* __restrict__ out, int n) {
    const int t = blockIdx.x * blockDim.x + threadIdx.x;
    if (t < n) out[t] = __float2bfloat16(in[t]);
}

// ---------------------------------------------------------------------------
// bf16 GEMM: C[M,N] = A[M,K] * Bt[N,K]^T  (both row-major, Bt is W transposed)
// 128 threads = 4 waves; wave tile 64x64 (4x4 WMMA); block tile 128x128.
// Per k-step: 8 fragment loads feed 16 WMMAs (2 WMMA per 32B load).
// ---------------------------------------------------------------------------
template <typename OutT>
__global__ __launch_bounds__(128) void gemm_bf16_kernel(
    const __hip_bfloat16* __restrict__ A, const __hip_bfloat16* __restrict__ Bt,
    OutT* __restrict__ C, int M, int N, int K) {
    const int lane = threadIdx.x & 31;
    const int wave = threadIdx.x >> 5;              // 0..3
    const int m0 = blockIdx.y * 128 + (wave >> 1) * 64;
    const int n0 = blockIdx.x * 128 + (wave & 1) * 64;
    const int half = lane >> 4, lr = lane & 15, khalf = half * 8;

    const __hip_bfloat16* arow[4];
    const __hip_bfloat16* brow[4];
    #pragma unroll
    for (int i = 0; i < 4; ++i) {
        arow[i] = A + (size_t)(m0 + i * 16 + lr) * K;
        brow[i] = Bt + (size_t)(n0 + i * 16 + lr) * K;
    }
    v8f acc[4][4];
    #pragma unroll
    for (int i = 0; i < 4; ++i)
        #pragma unroll
        for (int j = 0; j < 4; ++j) acc[i][j] = {};

    for (int kk = 0; kk < K; kk += 32) {
        __builtin_prefetch(arow[0] + kk + 256, 0, 0);   // global_prefetch_b8
        __builtin_prefetch(brow[0] + kk + 256, 0, 0);
        v16bf av[4], bv[4];
        #pragma unroll
        for (int i = 0; i < 4; ++i) av[i] = load_frag(arow[i] + kk, khalf);
        #pragma unroll
        for (int j = 0; j < 4; ++j) bv[j] = load_frag(brow[j] + kk, khalf);
        #pragma unroll
        for (int i = 0; i < 4; ++i)
            #pragma unroll
            for (int j = 0; j < 4; ++j)
                acc[i][j] = wmma_bf16(av[i], bv[j], acc[i][j]);
    }
    // C/D layout: VGPR r holds (row = r + 8*(lane>>4), col = lane&15)
    #pragma unroll
    for (int i = 0; i < 4; ++i)
        #pragma unroll
        for (int j = 0; j < 4; ++j)
            #pragma unroll
            for (int r = 0; r < 8; ++r) {
                const int row = m0 + i * 16 + r + 8 * half;
                const int col = n0 + j * 16 + lr;
                C[(size_t)row * N + col] = to_out<OutT>(acc[i][j][r]);
            }
}

// ---------------------------------------------------------------------------
// RoPE (in-place, bf16 storage, f32 math). X layout [B*S, heads*64].
// One thread per (token, head, i<32): rotates (d=i, d=i+32).
// ---------------------------------------------------------------------------
__global__ void rope_kernel(__hip_bfloat16* __restrict__ X, int heads, int total) {
    const int t = blockIdx.x * blockDim.x + threadIdx.x;
    if (t >= total) return;
    const int i  = t & 31;
    const int th = t >> 5;
    const int hh = th % heads;
    const int bs = th / heads;
    const int s  = bs & (S_ - 1);
    const float f = __expf(-(float)i * 0.2878231366f);   // theta^(-2i/64), ln(1e4)/32
    const float ang = (float)s * f;
    float sn, c;
    __sincosf(ang, &sn, &c);
    __hip_bfloat16* p = X + (size_t)bs * (heads * HD_) + hh * HD_;
    const float x0 = __bfloat162float(p[i]);
    const float x1 = __bfloat162float(p[i + 32]);
    p[i]      = __float2bfloat16(x0 * c - x1 * sn);
    p[i + 32] = __float2bfloat16(x1 * c + x0 * sn);
}

// V[B*S, 512] -> Vt[B, NKV, HD, S]  (columns of V become contiguous over s)
__global__ void transpose_v_kernel(const __hip_bfloat16* __restrict__ V,
                                   __hip_bfloat16* __restrict__ Vt, int total) {
    const int t = blockIdx.x * blockDim.x + threadIdx.x;
    if (t >= total) return;
    const int d  = t & 63;
    const int hh = (t >> 6) & 7;
    const int bs = t >> 9;
    const int b  = bs >> 11;          // / 2048
    const int s  = bs & 2047;
    Vt[(((size_t)b * NKV_ + hh) * HD_ + d) * S_ + s] = V[t];
}

// ---------------------------------------------------------------------------
// Flash attention (causal, GQA). One wave per (b, h, 16-query tile).
// Key tiles of 32. Scores & PV via bf16 WMMA, online softmax in f32.
// Row max via DPP16 butterfly; row sum via WMMA against a ones fragment
// (the running denominator lacc is rescaled exactly like the O accumulators).
// ---------------------------------------------------------------------------
template <bool MASKED>
__device__ __forceinline__ void attn_tile(
    int kt, int b, int qt, int h, int kvh, int lane,
    const __hip_bfloat16* __restrict__ Kc, const __hip_bfloat16* __restrict__ Vt,
    __hip_bfloat16* ldsPw, const v16bf& onesv,
    const v16bf& aq0, const v16bf& aq1,
    v8f& o0, v8f& o1, v8f& o2, v8f& o3, v8f& lacc,
    float (&mrow)[8]) {
    const int half = lane >> 4, lr = lane & 15, khalf = half * 8;
    const int kbase = kt * 32;
    v8f s0 = {}, s1 = {};
    {   // kpos sub-tile 0..15 : lane = kpos column, contiguous over d
        const __hip_bfloat16* krow = Kc + ((size_t)b * S_ + kbase + lr) * KV_ + kvh * HD_;
        s0 = wmma_bf16(aq0, load_frag(krow, khalf), s0);
        s0 = wmma_bf16(aq1, load_frag(krow + 32, khalf), s0);
    }
    {   // kpos sub-tile 16..31
        const __hip_bfloat16* krow = Kc + ((size_t)b * S_ + kbase + 16 + lr) * KV_ + kvh * HD_;
        s1 = wmma_bf16(aq0, load_frag(krow, khalf), s1);
        s1 = wmma_bf16(aq1, load_frag(krow + 32, khalf), s1);
    }
    // online softmax: scale, (mask), per-row max via DPP16 butterfly
    #pragma unroll
    for (int r = 0; r < 8; ++r) {
        float v0 = s0[r] * 0.125f;              // 1/sqrt(64)
        float v1 = s1[r] * 0.125f;
        if (MASKED) {
            const int qg = qt * 16 + r + 8 * half;
            if (kbase + lr > qg)      v0 = -3.0e38f;
            if (kbase + 16 + lr > qg) v1 = -3.0e38f;
        }
        const float mx = rowmax16_dpp(fmaxf(v0, v1));
        const float mnew = fmaxf(mrow[r], mx);
        const float corr = __expf(mrow[r] - mnew);
        const float p0 = __expf(v0 - mnew);
        const float p1 = __expf(v1 - mnew);
        mrow[r] = mnew;
        o0[r] *= corr; o1[r] *= corr; o2[r] *= corr; o3[r] *= corr;
        lacc[r] *= corr;
        const int prow = r + 8 * half;          // stage P in A-fragment layout
        ldsPw[prow * 32 + lr]      = __float2bfloat16(p0);
        ldsPw[prow * 32 + 16 + lr] = __float2bfloat16(p1);
    }
    asm volatile("s_wait_dscnt 0" ::: "memory");
    FragU pf;   // P A-fragment: lane = query row lr, K = kpos
    const uint4* lp = reinterpret_cast<const uint4*>(ldsPw + lr * 32);
    pf.u[0] = lp[half];
    pf.u[1] = lp[half + 2];
    #pragma unroll
    for (int t = 0; t < 4; ++t) {   // O[q, d] += P * V ; lane = d column
        const __hip_bfloat16* vcol =
            Vt + (((size_t)b * NKV_ + kvh) * HD_ + t * 16 + lr) * S_ + kbase;
        const v16bf bv = load_frag(vcol, khalf);
        v8f* oc = (t == 0) ? &o0 : (t == 1) ? &o1 : (t == 2) ? &o2 : &o3;
        *oc = wmma_bf16(pf.v, bv, *oc);
    }
    lacc = wmma_bf16(pf.v, onesv, lacc);    // row sums of P, every lane/column
}

__global__ __launch_bounds__(256) void attn_kernel(
    const __hip_bfloat16* __restrict__ Q,   // [B*S, 2048]
    const __hip_bfloat16* __restrict__ Kc,  // [B*S, 512]
    const __hip_bfloat16* __restrict__ Vt,  // [B, NKV, HD, S]
    __hip_bfloat16* __restrict__ O) {       // [B*S, 2048]
    __shared__ __align__(16) __hip_bfloat16 ldsP[8][16 * 32];
    const int lane = threadIdx.x & 31;
    const int wave = threadIdx.x >> 5;
    const int task = blockIdx.x * 8 + wave;     // 0..8191
    const int qt = task & 127;
    const int h  = (task >> 7) & 31;
    const int b  = task >> 12;
    const int kvh = h >> 2;                     // GROUPS = 4
    const int half = lane >> 4, lr = lane & 15, khalf = half * 8;

    // constant all-ones bf16 B fragment (0x3F80 per element)
    FragU onesf;
    onesf.u[0] = make_uint4(0x3F803F80u, 0x3F803F80u, 0x3F803F80u, 0x3F803F80u);
    onesf.u[1] = onesf.u[0];

    // Q A-fragments for this query tile (rows = lr), d-chunks {0..31},{32..63}
    const __hip_bfloat16* qrow = Q + ((size_t)b * S_ + (size_t)qt * 16 + lr) * H_ + h * HD_;
    const v16bf aq0 = load_frag(qrow, khalf);
    const v16bf aq1 = load_frag(qrow + 32, khalf);

    v8f o0 = {}, o1 = {}, o2 = {}, o3 = {}, lacc = {};
    float mrow[8];
    #pragma unroll
    for (int r = 0; r < 8; ++r) mrow[r] = -3.0e38f;

    const int ktfull = (qt * 16 + 1) >> 5;      // tiles with kmax <= qmin: no mask
    const int ktmax  = (qt * 16 + 15) >> 5;     // last (diagonal) tile, inclusive
    int kt = 0;
    for (; kt < ktfull; ++kt)
        attn_tile<false>(kt, b, qt, h, kvh, lane, Kc, Vt, ldsP[wave], onesf.v,
                         aq0, aq1, o0, o1, o2, o3, lacc, mrow);
    for (; kt <= ktmax; ++kt)
        attn_tile<true>(kt, b, qt, h, kvh, lane, Kc, Vt, ldsP[wave], onesf.v,
                        aq0, aq1, o0, o1, o2, o3, lacc, mrow);

    #pragma unroll
    for (int r = 0; r < 8; ++r) {
        const float inv = 1.0f / lacc[r];
        const int qg = qt * 16 + r + 8 * half;
        __hip_bfloat16* op = O + ((size_t)b * S_ + qg) * H_ + h * HD_;
        op[lr]      = __float2bfloat16(o0[r] * inv);
        op[16 + lr] = __float2bfloat16(o1[r] * inv);
        op[32 + lr] = __float2bfloat16(o2[r] * inv);
        op[48 + lr] = __float2bfloat16(o3[r] * inv);
    }
}

// ---------------------------------------------------------------------------
extern "C" void kernel_launch(void* const* d_in, const int* in_sizes, int n_in,
                              void* d_out, int out_size, void* d_ws, size_t ws_size,
                              hipStream_t stream) {
    (void)in_sizes; (void)n_in; (void)out_size; (void)ws_size;
    const float* hs   = (const float*)d_in[0];
    const float* bank = (const float*)d_in[1];
    const int* qidx = (const int*)d_in[2];
    const int* kidx = (const int*)d_in[3];
    const int* vidx = (const int*)d_in[4];
    const int* oidx = (const int*)d_in[5];
    float* out = (float*)d_out;

    char* ws = (char*)d_ws;
    size_t off = 0;
    auto alloc = [&](size_t bytes) -> char* {
        char* p = ws + off;
        off += (bytes + 255) & ~(size_t)255;
        return p;
    };
    __hip_bfloat16* hsb = (__hip_bfloat16*)alloc((size_t)BS_ * H_ * 2);   // 16 MB
    __hip_bfloat16* WqT = (__hip_bfloat16*)alloc((size_t)H_ * H_ * 2);    //  8 MB
    __hip_bfloat16* WkT = (__hip_bfloat16*)alloc((size_t)KV_ * H_ * 2);   //  2 MB
    __hip_bfloat16* WvT = (__hip_bfloat16*)alloc((size_t)KV_ * H_ * 2);   //  2 MB
    __hip_bfloat16* WoT = (__hip_bfloat16*)alloc((size_t)H_ * H_ * 2);    //  8 MB
    __hip_bfloat16* Qb  = (__hip_bfloat16*)alloc((size_t)BS_ * H_ * 2);   // 16 MB
    __hip_bfloat16* Kb  = (__hip_bfloat16*)alloc((size_t)BS_ * KV_ * 2);  //  4 MB
    __hip_bfloat16* Vb  = (__hip_bfloat16*)alloc((size_t)BS_ * KV_ * 2);  //  4 MB
    __hip_bfloat16* Vtb = (__hip_bfloat16*)alloc((size_t)BS_ * KV_ * 2);  //  4 MB
    __hip_bfloat16* Ob  = (__hip_bfloat16*)alloc((size_t)BS_ * H_ * 2);   // 16 MB

    // 1) gather codebook weights (transposed, bf16) + cast hidden states
    {
        const size_t tq = (size_t)H_ * H_;
        gather_wT_kernel<<<(unsigned)((tq + 255) / 256), 256, 0, stream>>>(bank, qidx, WqT, H_, H_);
        const size_t tk = (size_t)H_ * KV_;
        gather_wT_kernel<<<(unsigned)((tk + 255) / 256), 256, 0, stream>>>(bank, kidx, WkT, H_, KV_);
        gather_wT_kernel<<<(unsigned)((tk + 255) / 256), 256, 0, stream>>>(bank, vidx, WvT, H_, KV_);
        gather_wT_kernel<<<(unsigned)((tq + 255) / 256), 256, 0, stream>>>(bank, oidx, WoT, H_, H_);
        const int th = BS_ * H_;
        cast_bf16_kernel<<<(th + 255) / 256, 256, 0, stream>>>(hs, hsb, th);
    }
    // 2) projections (bf16 WMMA GEMM, 128x128 block / 64x64 wave tiles)
    gemm_bf16_kernel<__hip_bfloat16><<<dim3(H_ / 128, BS_ / 128), 128, 0, stream>>>(hsb, WqT, Qb, BS_, H_, H_);
    gemm_bf16_kernel<__hip_bfloat16><<<dim3(KV_ / 128, BS_ / 128), 128, 0, stream>>>(hsb, WkT, Kb, BS_, KV_, H_);
    gemm_bf16_kernel<__hip_bfloat16><<<dim3(KV_ / 128, BS_ / 128), 128, 0, stream>>>(hsb, WvT, Vb, BS_, KV_, H_);
    // 3) RoPE on Q and K
    {
        const int tq = BS_ * NH_ * 32;
        rope_kernel<<<(tq + 255) / 256, 256, 0, stream>>>(Qb, NH_, tq);
        const int tk = BS_ * NKV_ * 32;
        rope_kernel<<<(tk + 255) / 256, 256, 0, stream>>>(Kb, NKV_, tk);
    }
    // 4) transpose V for contiguous PV B-fragments
    {
        const int tv = BS_ * KV_;
        transpose_v_kernel<<<(tv + 255) / 256, 256, 0, stream>>>(Vb, Vtb, tv);
    }
    // 5) flash attention: 8192 wave tasks, 8 waves/block
    attn_kernel<<<1024, 256, 0, stream>>>(Qb, Kb, Vtb, Ob);
    // 6) output projection -> f32 d_out
    gemm_bf16_kernel<float><<<dim3(H_ / 128, BS_ / 128), 128, 0, stream>>>(Ob, WoT, out, BS_, H_, H_);
}
